// ParametricDist_54743653155010
// MI455X (gfx1250) — matrix operand order repaired and checked by reading, not compiled, not run on hardware
//
#include <hip/hip_runtime.h>

#define DIN   1600
#define NROWS 1024
#define MROWS 1024
#define H1D   64

typedef __attribute__((ext_vector_type(16))) _Float16 v16h;
typedef __attribute__((ext_vector_type(8)))  float    v8f;

// ---------------------------------------------------------------------------
// Kernel 1: pah = f16(a @ W1^T + b1)  (1024x64), pbh = f16(b @ W1^T) (1024x64).
// One thread per output element; 400 float4 FMAs each (f32 accumulate),
// single f16 convert at the store. ~0.4 GFLOP total.
// ---------------------------------------------------------------------------
__global__ void __launch_bounds__(256) pd_proj(
    const float* __restrict__ a, const float* __restrict__ b,
    const float* __restrict__ W1, const float* __restrict__ b1,
    _Float16* __restrict__ pah, _Float16* __restrict__ pbh) {
  int t   = blockIdx.x * blockDim.x + threadIdx.x;   // 0 .. 2*1024*64-1
  int h   = t & (H1D - 1);
  int row = t >> 6;
  const float* src = (row < NROWS) ? (a + (size_t)row * DIN)
                                   : (b + (size_t)(row - NROWS) * DIN);
  const float4* sv = (const float4*)src;
  const float4* wv = (const float4*)(W1 + (size_t)h * DIN);
  float s0 = 0.f, s1 = 0.f, s2 = 0.f, s3 = 0.f;
#pragma unroll 4
  for (int i = 0; i < DIN / 4; ++i) {
    float4 x = sv[i];
    float4 y = wv[i];
    s0 += x.x * y.x;
    s1 += x.y * y.y;
    s2 += x.z * y.z;
    s3 += x.w * y.w;
  }
  float sum = (s0 + s1) + (s2 + s3);
  if (row < NROWS) {
    pah[row * H1D + h] = (_Float16)(sum + b1[h]);   // fold layer-1 bias into pa
  } else {
    pbh[(row - NROWS) * H1D + h] = (_Float16)sum;
  }
}

// ---------------------------------------------------------------------------
// Kernel 2: fused pair MLP, one n x 64 m's per wave (4 tiles of 16 m's).
// Layer 1 in packed f16: bf = v_pk_max(v_pk_sub(c, pb), 0).
// Layer 2 as D = W2 x h1^T via v_wmma_f32_16x16x32_f16:
//   A (16x32 f16) = W2 feature rows   (loop-invariant, built once per wave)
//   B (32x16 f16) = h1^T (K=features, N=m's), contiguous 16-half runs
//   C/D layout: lane%16 = m column, vgpr + 8*(lane/16) = feature row
// Layer 3: register dot over this lane's 16 features + one shfl_xor(16),
// then 16 coalesced f32 stores per tile.
// ---------------------------------------------------------------------------
__global__ void __launch_bounds__(128) pd_main(
    const _Float16* __restrict__ pah, const _Float16* __restrict__ pbh,
    const float* __restrict__ W2, const float* __restrict__ b2,
    const float* __restrict__ W3, const float* __restrict__ b3,
    float* __restrict__ out) {
  const int lane   = threadIdx.x & 31;
  const int halfId = lane >> 4;   // 0 or 1
  const int mrow   = lane & 15;   // A row (feature%16) / B column (m%16)
  const int wave   = blockIdx.x * 4 + (threadIdx.x >> 5);
  const int n      = wave >> 4;          // 16 m-tiles-of-64 per n row
  const int m0     = (wave & 15) << 6;   // first of 64 m's for this wave

  // ---- A fragments: W2 rows, contiguous 16-half K runs (built once) ----
  // aw[ft][kc]: lane holds feature row ft*16 + mrow;
  // elems 0-15 -> K kb+16*kc+0..15 (kb = half*16), contiguous per chunk.
  const int kb = halfId * 16;
  v16h aw[2][2];
#pragma unroll
  for (int ft = 0; ft < 2; ++ft) {
    const float* wrow = W2 + (size_t)(ft * 16 + mrow) * H1D;
#pragma unroll
    for (int r = 0; r < 4; ++r) {
      const int k = kb + 8 * r;               // r<2 -> K-chunk 0, r>=2 -> chunk 1
      float4 w0 = *(const float4*)(wrow + k);
      float4 w1 = *(const float4*)(wrow + k + 4);
      const int kc = r >> 1;
      const int o  = 8 * (r & 1);
      aw[ft][kc][o + 0] = (_Float16)w0.x;  aw[ft][kc][o + 1] = (_Float16)w0.y;
      aw[ft][kc][o + 2] = (_Float16)w0.z;  aw[ft][kc][o + 3] = (_Float16)w0.w;
      aw[ft][kc][o + 4] = (_Float16)w1.x;  aw[ft][kc][o + 5] = (_Float16)w1.y;
      aw[ft][kc][o + 6] = (_Float16)w1.z;  aw[ft][kc][o + 7] = (_Float16)w1.w;
    }
  }

  // ---- c = pa[n] + b1 (already folded), f16, at this lane's B k-pattern ----
  // B fragment chunk ch: halves i=0..15 -> K = ch*32 + halfId*16 + i.
  const _Float16* paRow = pah + (size_t)n * H1D;
  v16h cbh[2];
  cbh[0] = *(const v16h*)(paRow + halfId * 16);
  cbh[1] = *(const v16h*)(paRow + 32 + halfId * 16);

  // ---- per-lane W3 / b2 slices matching the accumulator layout ----
  // acc[ft][v] = feature ft*16 + halfId*8 + v  for this lane.
  float w3s[2][8], b2s[2][8];
#pragma unroll
  for (int ft = 0; ft < 2; ++ft) {
    const int base = ft * 16 + halfId * 8;
    float4 u0 = *(const float4*)(W3 + base);
    float4 u1 = *(const float4*)(W3 + base + 4);
    float4 v0 = *(const float4*)(b2 + base);
    float4 v1 = *(const float4*)(b2 + base + 4);
    w3s[ft][0] = u0.x; w3s[ft][1] = u0.y; w3s[ft][2] = u0.z; w3s[ft][3] = u0.w;
    w3s[ft][4] = u1.x; w3s[ft][5] = u1.y; w3s[ft][6] = u1.z; w3s[ft][7] = u1.w;
    b2s[ft][0] = v0.x; b2s[ft][1] = v0.y; b2s[ft][2] = v0.z; b2s[ft][3] = v0.w;
    b2s[ft][4] = v1.x; b2s[ft][5] = v1.y; b2s[ft][6] = v1.z; b2s[ft][7] = v1.w;
  }
  const float bias3 = b3[0];
  const v16h  zeroh = {};

#pragma unroll
  for (int mt = 0; mt < 4; ++mt) {
    const int mbase = m0 + mt * 16;

    // ---- Layer 1 (packed f16): bf = max(c - pb[m], 0), m = mbase + mrow ----
    const _Float16* pbRow = pbh + (size_t)(mbase + mrow) * H1D;
    v16h p0 = *(const v16h*)(pbRow + halfId * 16);
    v16h p1 = *(const v16h*)(pbRow + 32 + halfId * 16);
    v16h bf0 = __builtin_elementwise_max(cbh[0] - p0, zeroh);
    v16h bf1 = __builtin_elementwise_max(cbh[1] - p1, zeroh);

    // ---- Layer 2: h2_pre[m, feat] via WMMA, acc0/acc1 interleaved ----
    v8f acc0 = {0.f, 0.f, 0.f, 0.f, 0.f, 0.f, 0.f, 0.f};  // features 0..15
    v8f acc1 = {0.f, 0.f, 0.f, 0.f, 0.f, 0.f, 0.f, 0.f};  // features 16..31
    acc0 = __builtin_amdgcn_wmma_f32_16x16x32_f16(false, aw[0][0], false, bf0,
                                                  (short)0, acc0, false, false);
    acc1 = __builtin_amdgcn_wmma_f32_16x16x32_f16(false, aw[1][0], false, bf0,
                                                  (short)0, acc1, false, false);
    acc0 = __builtin_amdgcn_wmma_f32_16x16x32_f16(false, aw[0][1], false, bf1,
                                                  (short)0, acc0, false, false);
    acc1 = __builtin_amdgcn_wmma_f32_16x16x32_f16(false, aw[1][1], false, bf1,
                                                  (short)0, acc1, false, false);

    // ---- Layer 3: register dot over this lane's 16 features + 1 shuffle ----
    float s = 0.f;
#pragma unroll
    for (int v = 0; v < 8; ++v) {
      s += w3s[0][v] * fmaxf(acc0[v] + b2s[0][v], 0.f);
      s += w3s[1][v] * fmaxf(acc1[v] + b2s[1][v], 0.f);
    }
    s += __shfl_xor(s, 16, 32);   // combine the two half-wave feature slices
    if (lane < 16) {
      out[(size_t)n * MROWS + mbase + lane] = fmaxf(s + bias3, 0.f);
    }
  }
}

// ---------------------------------------------------------------------------
// Launch: inputs in setup_inputs() order: a, b, W1, b1, W2, b2, W3, b3.
// Workspace: pah (1024*64 f16) then pbh (1024*64 f16) = 256 KB.
// ---------------------------------------------------------------------------
extern "C" void kernel_launch(void* const* d_in, const int* in_sizes, int n_in,
                              void* d_out, int out_size, void* d_ws, size_t ws_size,
                              hipStream_t stream) {
  const float* a  = (const float*)d_in[0];
  const float* b  = (const float*)d_in[1];
  const float* W1 = (const float*)d_in[2];
  const float* b1 = (const float*)d_in[3];
  const float* W2 = (const float*)d_in[4];
  const float* b2 = (const float*)d_in[5];
  const float* W3 = (const float*)d_in[6];
  const float* b3 = (const float*)d_in[7];
  float* out = (float*)d_out;

  _Float16* pah = (_Float16*)d_ws;
  _Float16* pbh = pah + (size_t)NROWS * H1D;

  // Projections: 2*1024*64 = 131072 threads.
  pd_proj<<<(2 * NROWS * H1D) / 256, 256, 0, stream>>>(a, b, W1, b1, pah, pbh);

  // Fused pair MLP: 1024 n * 16 m-tiles-of-64 = 16384 waves / 4 per block.
  pd_main<<<(NROWS * (MROWS / 64)) / 4, 128, 0, stream>>>(pah, pbh, W2, b2, W3, b3, out);
}
